// CustomModel_89008902243063
// MI455X (gfx1250) — compile-verified, matrix operand order checked by reading, not run on hardware
//
#include <hip/hip_runtime.h>

// ---------------- types ----------------
typedef __attribute__((ext_vector_type(16))) __bf16 bf16x16;
typedef __attribute__((ext_vector_type(8)))  __bf16 bf16x8;
typedef __attribute__((ext_vector_type(8)))  float  f32x8;
typedef __attribute__((ext_vector_type(4)))  unsigned int u32x4;
typedef unsigned short u16;

union BF16Frag  { bf16x16 v; u16 u[16]; u32x4 q[2]; bf16x8 h[2]; };
union Pack8     { u16 t[8]; u32x4 q; };
union BFBits    { __bf16 h; u16 u; };

__device__ __forceinline__ u16 f2bf(float f){
  BFBits cv; cv.h = (__bf16)f;            // native v_cvt (RTNE)
  return cv.u;
}
__device__ __forceinline__ float bf2f(u16 h){ return __uint_as_float(((unsigned)h) << 16); }

// load 16 bf16 fragment elements: 8 contiguous at +half*8, 8 at +16+half*8
__device__ __forceinline__ bf16x16 frag_ld(const u16* rowk, int half){
  BF16Frag r;
  r.q[0] = *(const u32x4*)(rowk + half*8);
  r.q[1] = *(const u32x4*)(rowk + 16 + half*8);
  return r.v;
}
// same but converting from fp32 source on the fly (32B-aligned rows)
__device__ __forceinline__ bf16x16 frag_ld_f32(const float* rowk, int half){
  BF16Frag r;
  f32x8 lo = *(const f32x8*)(rowk + half*8);
  f32x8 hi = *(const f32x8*)(rowk + 16 + half*8);
  r.h[0] = __builtin_convertvector(lo, bf16x8);   // packed v_cvt_pk_bf16_f32
  r.h[1] = __builtin_convertvector(hi, bf16x8);
  return r.v;
}
__device__ __forceinline__ f32x8 wmma_bf16(bf16x16 a, bf16x16 b, f32x8 c){
  return __builtin_amdgcn_wmma_f32_16x16x32_bf16(false, a, false, b, (short)0, c, false, false);
}

// ---------------- workspace layout ----------------
constexpr size_t AL(size_t x){ return (x + 255) & ~(size_t)255; }
constexpr size_t OFF_W1    = 0;                                      // 32x160 bf16
constexpr size_t OFF_W2    = AL(OFF_W1   + 32*160*2);                // 32x288 bf16 (im2col reorder)
constexpr size_t OFF_W3    = AL(OFF_W2   + 32*288*2);                // 160x32 bf16
constexpr size_t OFF_WLIN  = AL(OFF_W3   + 160*32*2);                // 160x160 bf16
constexpr size_t OFF_WP2   = AL(OFF_WLIN + 160*160*2);               // 160x96 bf16 (proj_w[:,80:], K-padded)
constexpr size_t OFF_POS   = AL(OFF_WP2  + 160*96*2);                // 16129x4 f32
constexpr size_t OFF_SAT   = AL(OFF_POS  + (size_t)16129*4*4);       // 4x128x128 f32
constexpr size_t OFF_RPB   = AL(OFF_SAT  + (size_t)4*128*128*4);     // 4096x4 f32
constexpr size_t OFF_VPROJ = AL(OFF_RPB  + (size_t)4096*4*4);        // 32x80 f32
constexpr size_t OFF_CORR  = AL(OFF_VPROJ+ (size_t)32*80*4);         // 32x4x4096 f32
constexpr size_t OFF_TBUF  = AL(OFF_CORR + (size_t)32*4*4096*4);     // 32x160x4 f32
// ---- zero span (cleared every launch) ----
constexpr size_t OFF_H1T   = AL(OFF_TBUF + (size_t)32*160*4*4);      // 32 x (66*66) x 32 bf16 (padded planar-token)
constexpr size_t OFF_H2T   = AL(OFF_H1T  + (size_t)32*4356*32*2);    // 32 x 4096 x 32 bf16
constexpr size_t OFF_QVB   = AL(OFF_H2T  + (size_t)32*4096*32*2);    // 32 x 4096 x 192 bf16 (token-major, K-pad)
constexpr size_t OFF_CORRC = AL(OFF_QVB  + (size_t)32*4096*192*2);   // 32 x 80 x 96 bf16  (corr_c^T, K-pad)
constexpr size_t OFF_M2B   = AL(OFF_CORRC+ (size_t)32*80*96*2);      // 32 x 160 x 96 bf16 (M2, K-pad)
constexpr size_t OFF_ZEND  = AL(OFF_M2B  + (size_t)32*160*96*2);
// ---- end zero span ----
constexpr size_t OFF_QVT   = OFF_ZEND;                               // 32 x 160 x 4096 bf16 (channel-major)

// ---------------- kernels ----------------
__global__ void zero_kernel(u32x4* p, size_t n){
  size_t i = (size_t)blockIdx.x * blockDim.x + threadIdx.x;
  size_t s = (size_t)gridDim.x * blockDim.x;
  u32x4 z = (u32x4)0;
  for (; i < n; i += s) p[i] = z;
}

__global__ void prep_weights(const float* c1w, const float* c2w, const float* c3w,
                             const float* linw, const float* projw,
                             u16* w1b, u16* w2b, u16* w3b, u16* wlinb, u16* wp2b){
  for (int t = threadIdx.x; t < 5120;  t += 256) w1b[t]   = f2bf(c1w[t]);
  for (int t = threadIdx.x; t < 9216;  t += 256){
    int c = t / 288, r = t % 288, s = r >> 5, cp = r & 31;
    w2b[t] = f2bf(c2w[(c*32 + cp)*9 + s]);          // [c_out][(di*3+dj)*32 + c_in]
  }
  for (int t = threadIdx.x; t < 5120;  t += 256) w3b[t]   = f2bf(c3w[t]);
  for (int t = threadIdx.x; t < 25600; t += 256) wlinb[t] = f2bf(linw[t]);
  for (int t = threadIdx.x; t < 160*96; t += 256){
    int c = t / 96, e = t % 96;
    wp2b[t] = (e < 80) ? f2bf(projw[c*160 + 80 + e]) : (u16)0;
  }
}

__device__ __forceinline__ void ln_relu10(const float* t, const float* g, const float* b, float* tn){
  float m = 0.f;
  #pragma unroll
  for (int j = 0; j < 10; ++j) m += t[j];
  m *= 0.1f;
  float var = 0.f;
  #pragma unroll
  for (int j = 0; j < 10; ++j){ float d = t[j]-m; var += d*d; }
  var *= 0.1f;
  float inv = rsqrtf(var + 1e-5f);
  #pragma unroll
  for (int j = 0; j < 10; ++j){ float y = (t[j]-m)*inv*g[j] + b[j]; tn[j] = y > 0.f ? y : 0.f; }
}

__global__ void pos_mlp_kernel(const float* ppw, const float* ppb,
    const float* g1, const float* b1, const float* w1, const float* bb1,
    const float* g2, const float* b2, const float* w2, const float* bb2,
    const float* g3, const float* b3, const float* w3, const float* bb3,
    float* pos){
  int k = blockIdx.x*256 + threadIdx.x;
  if (k >= 127*127) return;
  float r0 = (float)(k/127 - 63), r1 = (float)(k%127 - 63);
  float t[10], tn[10], t2[10];
  #pragma unroll
  for (int j = 0; j < 10; ++j) t[j] = ppw[j*2]*r0 + ppw[j*2+1]*r1 + ppb[j];
  ln_relu10(t, g1, b1, tn);
  #pragma unroll
  for (int i = 0; i < 10; ++i){ float s = bb1[i];
    #pragma unroll
    for (int j = 0; j < 10; ++j) s += tn[j]*w1[i*10+j];
    t2[i] = s; }
  ln_relu10(t2, g2, b2, tn);
  #pragma unroll
  for (int i = 0; i < 10; ++i){ float s = bb2[i];
    #pragma unroll
    for (int j = 0; j < 10; ++j) s += tn[j]*w2[i*10+j];
    t[i] = s; }
  ln_relu10(t, g3, b3, tn);
  #pragma unroll
  for (int h = 0; h < 4; ++h){ float s = bb3[h];
    #pragma unroll
    for (int j = 0; j < 10; ++j) s += tn[j]*w3[h*10+j];
    pos[k*4+h] = s; }
}

// summed-area table over padded 128x128 per head
__global__ void sat_kernel(const float* pos, float* sat){
  int tid = threadIdx.x;
  for (int t = tid; t < 4*128; t += blockDim.x){
    int h = t >> 7, c = t & 127;
    sat[h*16384 + c] = 0.f;
    sat[h*16384 + c*128] = 0.f;
  }
  __syncthreads();
  for (int t = tid; t < 4*127; t += blockDim.x){
    int h = t / 127, r = t % 127 + 1;
    float run = 0.f;
    for (int c = 1; c <= 127; ++c){
      run += pos[((r-1)*127 + (c-1))*4 + h];
      sat[h*16384 + r*128 + c] = run;
    }
  }
  __syncthreads();
  for (int t = tid; t < 4*127; t += blockDim.x){
    int h = t / 127, c = t % 127 + 1;
    float acc = 0.f;
    for (int r = 1; r <= 127; ++r){
      acc += sat[h*16384 + r*128 + c];
      sat[h*16384 + r*128 + c] = acc;
    }
  }
}

__global__ void rpb_kernel(const float* sat, float* rpb){
  int t = blockIdx.x*256 + threadIdx.x;
  if (t >= 4096*4) return;
  int l = t >> 2, h = t & 3;
  int i = l >> 6, j = l & 63;
  const float* S = sat + h*16384;
  float s = S[(i+64)*128 + (j+64)] - S[i*128 + (j+64)] - S[(i+64)*128 + j] + S[i*128 + j];
  rpb[l*4 + h] = s * (1.f/4096.f);
}

// h1 = lrelu(c1 @ x)  ->  padded planar-token bf16 [b][pix66x66][32]
__global__ void conv1_kernel(const float* x, const u16* w1b, const float* c1b, u16* h1t){
  int wid = threadIdx.x >> 5, ln = threadIdx.x & 31;
  int g = blockIdx.x*8 + wid;
  int b = g >> 8, m0 = (g & 255) * 16;
  int half = ln >> 4, c0 = ln & 15;
  const float* arow = x + ((size_t)b*4096 + m0 + (ln & 15))*160;
  const u16* br0 = w1b + (size_t)c0*160;
  const u16* br1 = w1b + (size_t)(c0+16)*160;
  f32x8 a0 = (f32x8)0.f, a1 = (f32x8)0.f;
  #pragma unroll
  for (int k0 = 0; k0 < 160; k0 += 32){
    bf16x16 a = frag_ld_f32(arow + k0, half);
    a0 = wmma_bf16(a, frag_ld(br0 + k0, half), a0);
    a1 = wmma_bf16(a, frag_ld(br1 + k0, half), a1);
  }
  u16* hb = h1t + (size_t)b*4356*32;
  float bi0 = c1b[c0], bi1 = c1b[c0+16];
  #pragma unroll
  for (int v = 0; v < 8; ++v){
    int tok = m0 + v + 8*half;
    int pix = ((tok >> 6) + 1)*66 + (tok & 63) + 1;
    float y0 = a0[v] + bi0; y0 = y0 >= 0.f ? y0 : 0.2f*y0;
    float y1 = a1[v] + bi1; y1 = y1 >= 0.f ? y1 : 0.2f*y1;
    hb[(size_t)pix*32 + c0]      = f2bf(y0);
    hb[(size_t)pix*32 + c0 + 16] = f2bf(y1);
  }
}

// h2 = lrelu(3x3 conv SAME) via im2col WMMA (K = 9 * 32)
__global__ void conv2_kernel(const u16* h1t, const u16* w2b, const float* c2b, u16* h2t){
  int wid = threadIdx.x >> 5, ln = threadIdx.x & 31;
  int g = blockIdx.x*8 + wid;
  int b = g >> 8, m0 = (g & 255) * 16;
  int half = ln >> 4, c0 = ln & 15;
  int i = m0 >> 6, j = (m0 & 63) + (ln & 15);
  const u16* hb = h1t + (size_t)b*4356*32;
  const u16* br0 = w2b + (size_t)c0*288;
  const u16* br1 = w2b + (size_t)(c0+16)*288;
  f32x8 a0 = (f32x8)0.f, a1 = (f32x8)0.f;
  #pragma unroll
  for (int s = 0; s < 9; ++s){
    int pix = (i + s/3)*66 + (j + s%3);
    bf16x16 a = frag_ld(hb + (size_t)pix*32, half);
    a0 = wmma_bf16(a, frag_ld(br0 + s*32, half), a0);
    a1 = wmma_bf16(a, frag_ld(br1 + s*32, half), a1);
  }
  u16* ob = h2t + (size_t)b*4096*32;
  float bi0 = c2b[c0], bi1 = c2b[c0+16];
  #pragma unroll
  for (int v = 0; v < 8; ++v){
    int tok = m0 + v + 8*half;
    float y0 = a0[v] + bi0; y0 = y0 >= 0.f ? y0 : 0.2f*y0;
    float y1 = a1[v] + bi1; y1 = y1 >= 0.f ? y1 : 0.2f*y1;
    ob[(size_t)tok*32 + c0]      = f2bf(y0);
    ob[(size_t)tok*32 + c0 + 16] = f2bf(y1);
  }
}

// qv = (c3@h2 + b3) * (lin@x + blin); store token-major (192-pad) + channel-major
__global__ void conv3lin_kernel(const u16* h2t, const float* x, const u16* w3b, const u16* wlinb,
                                const float* c3b, const float* linb, u16* qvb, u16* qvT){
  int wid = threadIdx.x >> 5, ln = threadIdx.x & 31;
  int g = blockIdx.x*8 + wid;
  int b = g >> 8, m0 = (g & 255) * 16;
  int half = ln >> 4;
  const u16*  h2row = h2t + ((size_t)b*4096 + m0 + (ln & 15))*32;
  const float* xrow = x   + ((size_t)b*4096 + m0 + (ln & 15))*160;
  bf16x16 ah = frag_ld(h2row, half);
  bf16x16 ag[5];
  #pragma unroll
  for (int kk = 0; kk < 5; ++kk) ag[kk] = frag_ld_f32(xrow + kk*32, half);
  u16* qb = qvb + (size_t)b*4096*192;
  u16* qt = qvT + (size_t)b*160*4096;
  for (int nt = 0; nt < 10; ++nt){
    int c = nt*16 + (ln & 15);
    f32x8 acch = (f32x8)0.f, accg = (f32x8)0.f;
    acch = wmma_bf16(ah, frag_ld(w3b + (size_t)c*32, half), acch);
    #pragma unroll
    for (int kk = 0; kk < 5; ++kk)
      accg = wmma_bf16(ag[kk], frag_ld(wlinb + (size_t)c*160 + kk*32, half), accg);
    float b3 = c3b[c], bl = linb[c];
    Pack8 pk;
    #pragma unroll
    for (int v = 0; v < 8; ++v){
      int tok = m0 + v + 8*half;
      u16 hval = f2bf((acch[v] + b3) * (accg[v] + bl));
      qb[(size_t)tok*192 + c] = hval;
      pk.t[v] = hval;
    }
    *(u32x4*)(qt + (size_t)c*4096 + m0 + 8*half) = pk.q;  // 8 consecutive tokens
  }
}

__global__ void vproj_kernel(const u16* qvT, const float* slw, const float* slb, float* vproj){
  int b = blockIdx.x / 80, d = blockIdx.x % 80;
  const u16* row = qvT + ((size_t)b*160 + 80 + d)*4096;
  float s = 0.f;
  for (int n = threadIdx.x; n < 4096; n += 256) s += slw[n] * bf2f(row[n]);
  __shared__ float red[256];
  red[threadIdx.x] = s; __syncthreads();
  for (int st = 128; st > 0; st >>= 1){
    if (threadIdx.x < st) red[threadIdx.x] += red[threadIdx.x + st];
    __syncthreads();
  }
  if (threadIdx.x == 0) vproj[b*80 + d] = red[0] + slb[0];
}

__global__ void corr_kernel(const u16* qvb, const float* vproj, const float* rpb, float* corr){
  int t = blockIdx.x*256 + threadIdx.x;
  int b = t >> 12, l = t & 4095;
  const u16* q = qvb + ((size_t)b*4096 + l)*192;
  #pragma unroll
  for (int h = 0; h < 4; ++h){
    float s = 0.f;
    #pragma unroll
    for (int c = 0; c < 20; ++c) s += bf2f(q[h*20 + c]) * vproj[b*80 + h*20 + c];
    corr[((size_t)b*4 + h)*4096 + l] = s * 0.05f + rpb[l*4 + h];
  }
}

// corr_c^T[e][d] = (1/N) sum_l vf[l][e] qf[l][d]   (M=e rows from qvT v-part, N=d cols from q-part)
__global__ void corrc_kernel(const u16* qvT, u16* corrcb){
  int wid = threadIdx.x >> 5, ln = threadIdx.x & 31;
  int g = blockIdx.x*8 + wid;                 // 800 waves
  int b = g / 25, t = g % 25, mt = t / 5, nt = t % 5;
  int half = ln >> 4;
  const u16* arow = qvT + ((size_t)b*160 + 80 + mt*16 + (ln & 15))*4096;
  const u16* brow = qvT + ((size_t)b*160 +      nt*16 + (ln & 15))*4096;
  f32x8 acc = (f32x8)0.f;
  for (int k0 = 0; k0 < 4096; k0 += 32){
    __builtin_prefetch(arow + k0 + 512, 0, 1);   // global_prefetch_b8, stream ahead
    __builtin_prefetch(brow + k0 + 512, 0, 1);
    acc = wmma_bf16(frag_ld(arow + k0, half), frag_ld(brow + k0, half), acc);
  }
  u16* cb = corrcb + (size_t)b*80*96;
  int d = nt*16 + (ln & 15);
  #pragma unroll
  for (int v = 0; v < 8; ++v){
    int e = mt*16 + v + 8*half;
    cb[(size_t)e*96 + d] = f2bf(acc[v] * (1.f/4096.f));
  }
}

// M2[c][e] = sum_d Wp2[c][d] corr_c[d][e]
__global__ void m2_kernel(const u16* wp2b, const u16* corrcb, u16* m2b){
  int wid = threadIdx.x >> 5, ln = threadIdx.x & 31;
  int g = blockIdx.x*8 + wid;                 // 1600 waves
  int b = g / 50, t = g % 50, mt = t / 5, nt = t % 5;
  int half = ln >> 4;
  const u16* arow = wp2b  + (size_t)(mt*16 + (ln & 15))*96;
  const u16* brow = corrcb + (size_t)b*80*96 + (size_t)(nt*16 + (ln & 15))*96;
  f32x8 acc = (f32x8)0.f;
  #pragma unroll
  for (int k0 = 0; k0 < 96; k0 += 32)
    acc = wmma_bf16(frag_ld(arow + k0, half), frag_ld(brow + k0, half), acc);
  u16* mb = m2b + (size_t)b*160*96;
  int e = nt*16 + (ln & 15);
  #pragma unroll
  for (int v = 0; v < 8; ++v){
    int c = mt*16 + v + 8*half;
    mb[(size_t)c*96 + e] = f2bf(acc[v]);
  }
}

// T[b][c][h] = sum_{d in head h} vproj[b][d] * proj_w[c][d]
__global__ void tbuf_kernel(const float* vproj, const float* projw, float* tbuf){
  int t = blockIdx.x*256 + threadIdx.x;
  if (t >= 32*160) return;
  int b = t / 160, c = t % 160;
  #pragma unroll
  for (int h = 0; h < 4; ++h){
    float s = 0.f;
    #pragma unroll
    for (int cc = 0; cc < 20; ++cc) s += vproj[b*80 + h*20 + cc] * projw[(size_t)c*160 + h*20 + cc];
    tbuf[(size_t)t*4 + h] = s;
  }
}

// out[l][c] = proj_b[c] + sum_h corr[b,h,l]*T[b,c,h] + (vf @ M2^T)[l][c]
__global__ void out_kernel(const u16* qvb, const u16* m2b, const float* corr,
                           const float* tbuf, const float* projb, float* out){
  int wid = threadIdx.x >> 5, ln = threadIdx.x & 31;
  int g = blockIdx.x*8 + wid;                 // 81920 waves
  int b = g / 2560, r = g % 2560, mt = r / 10, nt = r % 10;
  int m0 = mt*16, half = ln >> 4;
  const u16* arow = qvb + ((size_t)b*4096 + m0 + (ln & 15))*192 + 80;   // vf, K padded to 96
  const u16* brow = m2b + (size_t)b*160*96 + (size_t)(nt*16 + (ln & 15))*96;
  f32x8 acc = (f32x8)0.f;
  #pragma unroll
  for (int k0 = 0; k0 < 96; k0 += 32)
    acc = wmma_bf16(frag_ld(arow + k0, half), frag_ld(brow + k0, half), acc);
  int c = nt*16 + (ln & 15);
  float th[4];
  #pragma unroll
  for (int h = 0; h < 4; ++h) th[h] = tbuf[((size_t)b*160 + c)*4 + h];
  float pb = projb[c];
  #pragma unroll
  for (int v = 0; v < 8; ++v){
    int tok = m0 + v + 8*half;
    float val = acc[v] + pb;
    #pragma unroll
    for (int h = 0; h < 4; ++h) val += corr[((size_t)b*4 + h)*4096 + tok] * th[h];
    out[((size_t)b*4096 + tok)*160 + c] = val;
  }
}

// ---------------- host ----------------
extern "C" void kernel_launch(void* const* d_in, const int* in_sizes, int n_in,
                              void* d_out, int out_size, void* d_ws, size_t ws_size,
                              hipStream_t stream){
  const float* x     = (const float*)d_in[0];
  const float* c1w   = (const float*)d_in[1];
  const float* c1b   = (const float*)d_in[2];
  const float* c2w   = (const float*)d_in[3];
  const float* c2b   = (const float*)d_in[4];
  const float* c3w   = (const float*)d_in[5];
  const float* c3b   = (const float*)d_in[6];
  const float* linw  = (const float*)d_in[7];
  const float* linb  = (const float*)d_in[8];
  const float* slw   = (const float*)d_in[9];
  const float* slb   = (const float*)d_in[10];
  const float* ppw   = (const float*)d_in[11];
  const float* ppb   = (const float*)d_in[12];
  const float* p1g   = (const float*)d_in[13];
  const float* p1b   = (const float*)d_in[14];
  const float* p1w   = (const float*)d_in[15];
  const float* p1bb  = (const float*)d_in[16];
  const float* p2g   = (const float*)d_in[17];
  const float* p2b   = (const float*)d_in[18];
  const float* p2w   = (const float*)d_in[19];
  const float* p2bb  = (const float*)d_in[20];
  const float* p3g   = (const float*)d_in[21];
  const float* p3b   = (const float*)d_in[22];
  const float* p3w   = (const float*)d_in[23];
  const float* p3bb  = (const float*)d_in[24];
  const float* projw = (const float*)d_in[25];
  const float* projb = (const float*)d_in[26];

  char* ws = (char*)d_ws;
  u16*   W1   = (u16*)(ws + OFF_W1);
  u16*   W2   = (u16*)(ws + OFF_W2);
  u16*   W3   = (u16*)(ws + OFF_W3);
  u16*   WLIN = (u16*)(ws + OFF_WLIN);
  u16*   WP2  = (u16*)(ws + OFF_WP2);
  float* POS  = (float*)(ws + OFF_POS);
  float* SAT  = (float*)(ws + OFF_SAT);
  float* RPB  = (float*)(ws + OFF_RPB);
  float* VPR  = (float*)(ws + OFF_VPROJ);
  float* CORR = (float*)(ws + OFF_CORR);
  float* TBUF = (float*)(ws + OFF_TBUF);
  u16*   H1T  = (u16*)(ws + OFF_H1T);
  u16*   H2T  = (u16*)(ws + OFF_H2T);
  u16*   QVB  = (u16*)(ws + OFF_QVB);
  u16*   CRCB = (u16*)(ws + OFF_CORRC);
  u16*   M2B  = (u16*)(ws + OFF_M2B);
  u16*   QVT  = (u16*)(ws + OFF_QVT);

  zero_kernel<<<4096, 256, 0, stream>>>((u32x4*)(ws + OFF_H1T), (OFF_ZEND - OFF_H1T)/16);
  prep_weights<<<1, 256, 0, stream>>>(c1w, c2w, c3w, linw, projw, W1, W2, W3, WLIN, WP2);
  pos_mlp_kernel<<<64, 256, 0, stream>>>(ppw, ppb, p1g, p1b, p1w, p1bb,
                                         p2g, p2b, p2w, p2bb, p3g, p3b, p3w, p3bb, POS);
  sat_kernel<<<1, 512, 0, stream>>>(POS, SAT);
  rpb_kernel<<<64, 256, 0, stream>>>(SAT, RPB);
  conv1_kernel<<<1024, 256, 0, stream>>>(x, W1, c1b, H1T);
  conv2_kernel<<<1024, 256, 0, stream>>>(H1T, W2, c2b, H2T);
  conv3lin_kernel<<<1024, 256, 0, stream>>>(H2T, x, W3, WLIN, c3b, linb, QVB, QVT);
  vproj_kernel<<<2560, 256, 0, stream>>>(QVT, slw, slb, VPR);
  corr_kernel<<<512, 256, 0, stream>>>(QVB, VPR, RPB, CORR);
  corrc_kernel<<<100, 256, 0, stream>>>(QVT, CRCB);
  m2_kernel<<<200, 256, 0, stream>>>(WP2, CRCB, M2B);
  tbuf_kernel<<<20, 256, 0, stream>>>(VPR, projw, TBUF);
  out_kernel<<<10240, 256, 0, stream>>>(QVB, M2B, CORR, TBUF, projb, (float*)d_out);

  (void)in_sizes; (void)n_in; (void)out_size; (void)ws_size;
}
